// FourierCrossAttentionW_1726576856560
// MI455X (gfx1250) — matrix operand order repaired and checked by reading, not compiled, not run on hardware
//
#include <hip/hip_runtime.h>
#include <hip/hip_bf16.h>

typedef __attribute__((ext_vector_type(16))) __bf16 v16bf;
typedef __attribute__((ext_vector_type(8)))  __bf16 v8bf;
typedef __attribute__((ext_vector_type(8)))  float  v8f;
typedef __attribute__((ext_vector_type(2)))  float  v2f;

#define B_   16
#define L_   2048
#define E_   64
#define H_   8
#define M_   64      // kept modes
#define BH_  128
#define TWO_PI_ 6.283185307179586f

// ---------------- workspace layout (bytes) ----------------
// FBfrag: bf16 [64 kchunk][8 strip][32 lane][16]  fwd basis, WMMA-B-frag order  @0x000000 (512 KB)
// IBfrag: bf16 [128 ct][4 kc][32 lane][16]        inv basis, WMMA-B-frag order  @0x080000 (512 KB)
// QKF   : f32  [2][BH][E][2M]  QF/KF spectra (re cols 0-63, im 64-127)          @0x100000 (8 MB)
// AT    : f32  [BH][2][64][64] tanh'ed attention matrix                         @0x900000 (4 MB)
#define WS_IB   (524288)
#define WS_QKF  (1048576)
#define WS_AT   (9437184)

#define AT_STRIDE 40      // LDS row stride (bf16) for K1 staging tile (80 B, 16B-mult)
#define OB_STRIDE 136     // LDS row stride (bf16) for K3 O spectrum (272 B, 16B-mult)

// numerically stable complex tanh: tanh(x+iy)
__device__ inline void ctanh_f32(float x, float y, float& re, float& im) {
    float a = 2.0f * x, b = 2.0f * y;
    float m  = __expf(-fabsf(a));              // e^{-|a|}
    float m2 = m * m;
    float cb = __cosf(b), sb = __sinf(b);
    float inv = 1.0f / (1.0f + m2 + 2.0f * m * cb);
    re = copysignf((1.0f - m2) * inv, a);
    im = 2.0f * m * sb * inv;
}

// ---------------- K0: build DFT bases directly in WMMA B-fragment order --------
__global__ __launch_bounds__(256) void fca_init_basis(__bf16* __restrict__ FBfrag,
                                                      __bf16* __restrict__ IBfrag) {
    int idx = blockIdx.x * blockDim.x + threadIdx.x;    // 0 .. 262143
    if (idx >= L_ * 2 * M_) return;
    int i    = idx & 15;
    int lane = (idx >> 4) & 31;
    int lhi  = lane >> 4, llo = lane & 15;
    // forward basis: [kchunk][strip][lane][16] ; elem i -> K=lhi*16+i, N=llo
    {
        int strip = (idx >> 9) & 7;
        int kc    = idx >> 12;                  // 0..63
        int l   = kc * 32 + lhi * 16 + i;       // time index
        int col = strip * 16 + llo;             // 0..127 : cos | -sin
        int m   = col & 63;
        int ph  = (l * m) & (L_ - 1);
        float ang = (float)ph * (TWO_PI_ / (float)L_);
        FBfrag[idx] = (__bf16)((col < 64) ? __cosf(ang) : -__sinf(ang));
    }
    // inverse basis: [ct][kc32][lane][16] ; elem i -> K=kc32*32+lhi*16+i, N=llo
    {
        int kc32 = (idx >> 9) & 3;
        int ct   = idx >> 11;                   // 0..127
        int row  = kc32 * 32 + lhi * 16 + i;    // 0..127 : cos rows | -sin rows
        int t    = ct * 16 + llo;
        int m    = row & 63;
        int ph   = (m * t) & (L_ - 1);
        float ang = (float)ph * (TWO_PI_ / (float)L_);
        float w = ((m == 0) ? 1.0f : 2.0f) * (1.0f / ((float)L_ * 4096.0f));
        IBfrag[idx] = (__bf16)((row < 64) ? w * __cosf(ang) : -w * __sinf(ang));
    }
}

// ---------------- K1: QF/KF = DFT of q/k  (M=64 x N=128 x K=2048, bf16 WMMA) ----
__global__ __launch_bounds__(256) void fca_fwd_dft(const float* __restrict__ q,
                                                   const float* __restrict__ k,
                                                   const __bf16* __restrict__ FBfrag,
                                                   float* __restrict__ QKF) {
    const int bh = blockIdx.x, src = blockIdx.y;
    const int b = bh >> 3, h = bh & 7;
    const float* __restrict__ x = src ? k : q;
    const int tid  = threadIdx.x;
    const int lane = tid & 31, wave = tid >> 5;
    const int lhi  = lane >> 4, llo = lane & 15;
    const int n0   = wave * 16;                 // each wave owns one 16-col strip

    __shared__ __align__(16) __bf16 At[E_ * AT_STRIDE];  // 64 e-rows, padded

    v8f acc[4] = {};
    for (int k0 = 0; k0 < L_; k0 += 32) {
        const int kc = k0 >> 5;
        if (k0 + 32 < L_)
            __builtin_prefetch((const void*)(FBfrag + ((size_t)(kc + 1) * 8 + wave) * 512),
                               0, 0);
        // stage strided [e,l] gather of q/k into LDS (f32 -> bf16)
        #pragma unroll
        for (int i = 0; i < 8; ++i) {
            int flat = tid * 8 + i;             // 0..2047
            int e = flat >> 5, kk = flat & 31;
            float v = x[(((size_t)b * L_ + (size_t)(k0 + kk)) * E_ + e) * H_ + h];
            At[e * AT_STRIDE + kk] = (__bf16)v;
        }
        __syncthreads();
        // B fragment: one coalesced 32-byte run per lane (pre-swizzled basis)
        v16bf bfr = *(const v16bf*)(FBfrag + (((size_t)kc * 8 + wave) * 32 + lane) * 16);
        #pragma unroll
        for (int mt = 0; mt < 4; ++mt) {
            // A fragment: two contiguous 8-element runs from LDS -> 2x ds_load_b128
            const __bf16* row = &At[(mt * 16 + llo) * AT_STRIDE + lhi * 16];
            v8bf lo = *(const v8bf*)(row);
            v8bf hi = *(const v8bf*)(row + 8);
            v16bf af;
            #pragma unroll
            for (int i = 0; i < 8; ++i) { af[i] = lo[i]; af[i + 8] = hi[i]; }
            acc[mt] = __builtin_amdgcn_wmma_f32_16x16x32_bf16(
                false, af, false, bfr, (short)0, acc[mt], false, false);
        }
        __syncthreads();
    }
    // store C tiles: VGPR r -> M = r + lhi*8, N = llo
    float* __restrict__ dst = QKF + (size_t)(src * BH_ + bh) * E_ * 128;
    #pragma unroll
    for (int mt = 0; mt < 4; ++mt)
        #pragma unroll
        for (int r = 0; r < 8; ++r) {
            int e = mt * 16 + r + lhi * 8;
            dst[(size_t)e * 128 + (n0 + llo)] = acc[mt][r];
        }
}

// ---------------- K2: A = tanh_c(QF^T x KF)  (64x64x64 complex, f32 WMMA) ------
__global__ __launch_bounds__(256) void fca_qk_tanh(const float* __restrict__ QKF,
                                                   float* __restrict__ AT) {
    const int bh = blockIdx.x;
    const int tid  = threadIdx.x;
    const int lane = tid & 31, wave = tid >> 5;
    const int lhi  = lane >> 4, llo = lane & 15;
    const float* __restrict__ QF = QKF + (size_t)bh * E_ * 128;
    const float* __restrict__ KF = QKF + (size_t)(BH_ + bh) * E_ * 128;
    float* __restrict__ ATre = AT + ((size_t)bh * 2 + 0) * 4096;
    float* __restrict__ ATim = AT + ((size_t)bh * 2 + 1) * 4096;

    #pragma unroll
    for (int t = 0; t < 2; ++t) {
        int tile = wave * 2 + t;
        int x0 = (tile >> 2) * 16, y0 = (tile & 3) * 16;
        v8f rr = {}, ii = {}, ri = {}, ir = {};
        for (int e0 = 0; e0 < E_; e0 += 4) {
            // f32 A frag (16x4): elem j -> M=llo, K=lhi*2+j ; B frag: N=llo, same K
            v2f aR, aI, bR, bI;
            #pragma unroll
            for (int jj = 0; jj < 2; ++jj) {
                int e = e0 + lhi * 2 + jj;
                aR[jj] = QF[(size_t)e * 128 + (x0 + llo)];
                aI[jj] = QF[(size_t)e * 128 + 64 + (x0 + llo)];
                bR[jj] = KF[(size_t)e * 128 + (y0 + llo)];
                bI[jj] = KF[(size_t)e * 128 + 64 + (y0 + llo)];
            }
            rr = __builtin_amdgcn_wmma_f32_16x16x4_f32(false, aR, false, bR, (short)0, rr, false, false);
            ii = __builtin_amdgcn_wmma_f32_16x16x4_f32(false, aI, false, bI, (short)0, ii, false, false);
            ri = __builtin_amdgcn_wmma_f32_16x16x4_f32(false, aR, false, bI, (short)0, ri, false, false);
            ir = __builtin_amdgcn_wmma_f32_16x16x4_f32(false, aI, false, bR, (short)0, ir, false, false);
        }
        #pragma unroll
        for (int r = 0; r < 8; ++r) {
            int xx = x0 + r + lhi * 8, yy = y0 + llo;
            float tre, tim;
            ctanh_f32(rr[r] - ii[r], ri[r] + ir[r], tre, tim);
            ATre[xx * 64 + yy] = tre;
            ATim[xx * 64 + yy] = tim;
        }
    }
}

// ---------------- K3: O = KF x A^T then inverse DFT to output -------------------
__global__ __launch_bounds__(256) void fca_out(const float* __restrict__ QKF,
                                               const float* __restrict__ AT,
                                               const __bf16* __restrict__ IBfrag,
                                               float* __restrict__ out) {
    const int bh = blockIdx.x, b = bh >> 3, h = bh & 7;
    const int tid  = threadIdx.x;
    const int lane = tid & 31, wave = tid >> 5;
    const int lhi  = lane >> 4, llo = lane & 15;
    const float* __restrict__ KF   = QKF + (size_t)(BH_ + bh) * E_ * 128;
    const float* __restrict__ ATre = AT + ((size_t)bh * 2 + 0) * 4096;
    const float* __restrict__ ATim = AT + ((size_t)bh * 2 + 1) * 4096;

    __shared__ __align__(16) __bf16 Ob[E_ * OB_STRIDE];  // [e][re 0-63 | im 64-127], padded

    // ---- phase A: O[e,x] = sum_y KF[e,y] * Atanh[x,y]  (complex, f32 WMMA) ----
    #pragma unroll
    for (int t = 0; t < 2; ++t) {
        int tile = wave * 2 + t;
        int m0 = (tile >> 2) * 16, x0 = (tile & 3) * 16;   // m0: e rows; x0: modes
        v8f rr = {}, ii = {}, ri = {}, ir = {};
        for (int y0 = 0; y0 < M_; y0 += 4) {
            v2f aR, aI, bR, bI;
            #pragma unroll
            for (int jj = 0; jj < 2; ++jj) {
                int y = y0 + lhi * 2 + jj;
                int e = m0 + llo;
                aR[jj] = KF[(size_t)e * 128 + y];
                aI[jj] = KF[(size_t)e * 128 + 64 + y];
                bR[jj] = ATre[(x0 + llo) * 64 + y];    // B[K=y, N=x] = A[x,y]
                bI[jj] = ATim[(x0 + llo) * 64 + y];
            }
            rr = __builtin_amdgcn_wmma_f32_16x16x4_f32(false, aR, false, bR, (short)0, rr, false, false);
            ii = __builtin_amdgcn_wmma_f32_16x16x4_f32(false, aI, false, bI, (short)0, ii, false, false);
            ri = __builtin_amdgcn_wmma_f32_16x16x4_f32(false, aR, false, bI, (short)0, ri, false, false);
            ir = __builtin_amdgcn_wmma_f32_16x16x4_f32(false, aI, false, bR, (short)0, ir, false, false);
        }
        #pragma unroll
        for (int r = 0; r < 8; ++r) {
            int e = m0 + r + lhi * 8, xx = x0 + llo;
            Ob[e * OB_STRIDE + xx]      = (__bf16)(rr[r] - ii[r]);
            Ob[e * OB_STRIDE + 64 + xx] = (__bf16)(ri[r] + ir[r]);
        }
    }
    __syncthreads();

    // ---- phase B: out[e,t] = [Ore|Oim] x [Ic;Is]   (64 x 2048 x 128, bf16 WMMA)
    for (int tt = 0; tt < 64; ++tt) {
        int g  = tt * 8 + wave;          // 512 tiles over 8 waves
        int mt = g & 3, ct = g >> 2;
        int m0 = mt * 16, t0 = ct * 16;
        v8f acc = {};
        #pragma unroll
        for (int kc = 0; kc < 128; kc += 32) {
            // A fragment: two contiguous 8-elem runs from padded LDS
            const __bf16* row = &Ob[(m0 + llo) * OB_STRIDE + kc + lhi * 16];
            v8bf lo = *(const v8bf*)(row);
            v8bf hi = *(const v8bf*)(row + 8);
            v16bf af;
            #pragma unroll
            for (int i = 0; i < 8; ++i) { af[i] = lo[i]; af[i + 8] = hi[i]; }
            // B fragment: coalesced pre-swizzled inverse basis
            v16bf bfr = *(const v16bf*)(IBfrag +
                (((size_t)ct * 4 + (kc >> 5)) * 32 + lane) * 16);
            acc = __builtin_amdgcn_wmma_f32_16x16x32_bf16(
                false, af, false, bfr, (short)0, acc, false, false);
        }
        #pragma unroll
        for (int r = 0; r < 8; ++r) {
            int e = m0 + r + lhi * 8, tp = t0 + llo;
            out[(((size_t)b * L_ + tp) * E_ + e) * H_ + h] = acc[r];
        }
    }
}

extern "C" void kernel_launch(void* const* d_in, const int* in_sizes, int n_in,
                              void* d_out, int out_size, void* d_ws, size_t ws_size,
                              hipStream_t stream) {
    (void)in_sizes; (void)n_in; (void)out_size; (void)ws_size;
    const float* q = (const float*)d_in[0];
    const float* k = (const float*)d_in[1];
    // d_in[2] = v (unused by the reference math), d_in[3] = mask (unused)
    float* out = (float*)d_out;
    char* ws = (char*)d_ws;
    __bf16* FBfrag = (__bf16*)(ws);
    __bf16* IBfrag = (__bf16*)(ws + WS_IB);
    float*  QKF    = (float*)(ws + WS_QKF);
    float*  AT     = (float*)(ws + WS_AT);

    fca_init_basis<<<1024, 256, 0, stream>>>(FBfrag, IBfrag);
    fca_fwd_dft<<<dim3(BH_, 2), 256, 0, stream>>>(q, k, FBfrag, QKF);
    fca_qk_tanh<<<BH_, 256, 0, stream>>>(QKF, AT);
    fca_out<<<BH_, 256, 0, stream>>>(QKF, AT, IBfrag, out);
}
